// MTResidualAdapter_35012573397442
// MI455X (gfx1250) — compile-verified, matrix operand order checked by reading, not run on hardware
//
#include <hip/hip_runtime.h>
#include <hip/hip_bf16.h>
#include <math.h>

// ---------------------------------------------------------------------------
// Problem dims (fixed by the reference)
// ---------------------------------------------------------------------------
constexpr int Bb   = 2;
constexpr int Ss   = 2048;
constexpr int Dd   = 512;
constexpr int Pp   = 13;
constexpr int Tt   = 5;
constexpr int Hh   = 64;
constexpr int NTOK = Bb * Ss;          // 4096 tokens
constexpr int NV   = Tt * Dd;          // 2560  (concatenated Wv columns)
constexpr int NCH  = 32;               // scan chunks
constexpr int CL   = Ss / NCH;         // 64 steps per chunk

typedef __bf16 bf16_t;
typedef __attribute__((ext_vector_type(16))) __bf16 v16bf;
typedef __attribute__((ext_vector_type(8)))  __bf16 v8bf;
typedef __attribute__((ext_vector_type(4)))  __bf16 v4bf;
typedef __attribute__((ext_vector_type(8)))  float  v8f;
typedef __attribute__((ext_vector_type(4)))  int    v4i;

union V16U { v16bf v; v8bf h[2]; };

// ---------------------------------------------------------------------------
// CDNA5 async global->LDS staging (ASYNCcnt-tracked), with safe fallback.
// Builtin signature (from hipcc diagnostic): param0 = v4i addrspace(1)*,
// param1 = v4i addrspace(3)*, then imm offset, imm cpol.
// ---------------------------------------------------------------------------
#if defined(__HIP_DEVICE_COMPILE__) && \
    __has_builtin(__builtin_amdgcn_global_load_async_to_lds_b128)
#define USE_ASYNC_LDS 1
#else
#define USE_ASYNC_LDS 0
#endif

#define GLOBAL_AS __attribute__((address_space(1)))
#define LDS_AS    __attribute__((address_space(3)))

__device__ __forceinline__ void lds_copy16(const bf16_t* g, bf16_t* l) {
#if USE_ASYNC_LDS
  __builtin_amdgcn_global_load_async_to_lds_b128(
      (GLOBAL_AS v4i*)g, (LDS_AS v4i*)l, 0, 0);
#else
  *(uint4*)l = *(const uint4*)g;
#endif
}

__device__ __forceinline__ void wait_stage() {
#if USE_ASYNC_LDS
#if __has_builtin(__builtin_amdgcn_s_wait_asynccnt)
  __builtin_amdgcn_s_wait_asynccnt(0);
#else
  asm volatile("s_wait_asynccnt 0x0" ::: "memory");
#endif
#endif
}

// ---------------------------------------------------------------------------
// Kernel 1: LayerNorm (fp32) -> xn (bf16, row-major 4096x512)
//           + gating MLP -> alpha[token][t]  (fp32)
// One wave per token, 8 tokens per 256-thread block.
// ---------------------------------------------------------------------------
__global__ __launch_bounds__(256) void k_ln_gate(
    const float* __restrict__ x, const float* __restrict__ gamma,
    const float* __restrict__ beta, const float* __restrict__ W1,
    const float* __restrict__ b1, const float* __restrict__ W2,
    const float* __restrict__ b2, const float* __restrict__ log_tau,
    bf16_t* __restrict__ xn_bf, float* __restrict__ alpha)
{
  __shared__ float xs[8][Dd];
  const int wave = threadIdx.x >> 5;
  const int lane = threadIdx.x & 31;
  const int token = blockIdx.x * 8 + wave;
  const float* row = x + (size_t)token * Dd;

  float4 v[4];
  float s1 = 0.f, s2 = 0.f;
#pragma unroll
  for (int i = 0; i < 4; ++i) {
    v[i] = ((const float4*)row)[lane + 32 * i];
    s1 += v[i].x + v[i].y + v[i].z + v[i].w;
    s2 += v[i].x * v[i].x + v[i].y * v[i].y + v[i].z * v[i].z + v[i].w * v[i].w;
  }
#pragma unroll
  for (int off = 16; off >= 1; off >>= 1) {
    s1 += __shfl_xor(s1, off);
    s2 += __shfl_xor(s2, off);
  }
  const float mu  = s1 * (1.f / Dd);
  const float var = s2 * (1.f / Dd) - mu * mu;
  const float inv = rsqrtf(var + 1e-5f);

#pragma unroll
  for (int i = 0; i < 4; ++i) {
    const int j = lane + 32 * i;               // float4 index, d = 4j..4j+3
    float4 g4 = ((const float4*)gamma)[j];
    float4 be4 = ((const float4*)beta)[j];
    float4 xn4;
    xn4.x = (v[i].x - mu) * inv * g4.x + be4.x;
    xn4.y = (v[i].y - mu) * inv * g4.y + be4.y;
    xn4.z = (v[i].z - mu) * inv * g4.z + be4.z;
    xn4.w = (v[i].w - mu) * inv * g4.w + be4.w;
    *(float4*)&xs[wave][4 * j] = xn4;
    v4bf o = { (bf16_t)xn4.x, (bf16_t)xn4.y, (bf16_t)xn4.z, (bf16_t)xn4.w };
    *(v4bf*)(xn_bf + (size_t)token * Dd + 4 * j) = o;
  }
  __syncthreads();

  // t1[h] = xn . W1[:,h] + b1[h]; each lane owns h = lane, lane+32
  float t1a = b1[lane];
  float t1b = b1[lane + 32];
#pragma unroll 8
  for (int k = 0; k < Dd; ++k) {
    const float xv = xs[wave][k];
    t1a = fmaf(xv, W1[k * Hh + lane], t1a);
    t1b = fmaf(xv, W1[k * Hh + lane + 32], t1b);
  }
  t1a = tanhf(t1a);
  t1b = tanhf(t1b);

  float gacc[Tt];
#pragma unroll
  for (int t = 0; t < Tt; ++t)
    gacc[t] = t1a * W2[lane * Tt + t] + t1b * W2[(lane + 32) * Tt + t];
#pragma unroll
  for (int off = 16; off >= 1; off >>= 1)
#pragma unroll
    for (int t = 0; t < Tt; ++t) gacc[t] += __shfl_xor(gacc[t], off);

  if (lane == 0) {
#pragma unroll
    for (int t = 0; t < Tt; ++t) {
      const float gate = 1.f / (1.f + expf(-(gacc[t] + b2[t])));
      alpha[(size_t)token * Tt + t] = expf(-gate * expf(-log_tau[t]));
    }
  }
}

// ---------------------------------------------------------------------------
// Kernel 2: fold mix over p into W_in:
//   WvT[t*512+d][k] = sum_p mix[p,t] * W_in[k][p*512+d]   (bf16, 2560x512)
//   bv[t*512+d]     = sum_p mix[p,t] * b_in[p*512+d]
// ---------------------------------------------------------------------------
__global__ __launch_bounds__(256) void k_build_wvt(
    const float* __restrict__ W_in, const float* __restrict__ b_in,
    const float* __restrict__ mix, bf16_t* __restrict__ WvT,
    float* __restrict__ bv)
{
  const int gid = blockIdx.x * 256 + threadIdx.x;   // 5*512*512 threads
  const int k = gid & 511;
  const int d = (gid >> 9) & 511;
  const int t = gid >> 18;
  float m[Pp];
#pragma unroll
  for (int p = 0; p < Pp; ++p) m[p] = mix[p * Tt + t];
  float s = 0.f;
#pragma unroll
  for (int p = 0; p < Pp; ++p)
    s = fmaf(m[p], W_in[(size_t)k * (Pp * Dd) + p * Dd + d], s);
  WvT[(size_t)(t * Dd + d) * Dd + k] = (bf16_t)s;
  if (k == 0) {
    float sb = 0.f;
#pragma unroll
    for (int p = 0; p < Pp; ++p) sb = fmaf(m[p], b_in[p * Dd + d], sb);
    bv[t * Dd + d] = sb;
  }
}

// Kernel 3: W_outT[n][k] = W_out[k][n]  (bf16)
__global__ __launch_bounds__(256) void k_build_woutT(
    const float* __restrict__ W_out, bf16_t* __restrict__ WoutT)
{
  const int gid = blockIdx.x * 256 + threadIdx.x;   // 512*512
  const int k = gid & 511;
  const int n = gid >> 9;
  WoutT[(size_t)n * Dd + k] = (bf16_t)W_out[(size_t)k * Dd + n];
}

// ---------------------------------------------------------------------------
// Generic bf16 WMMA GEMM:  C(MxN,f32) = epilogue( A(MxK) * BT(NxK)^T )
//   val = acc + bias[n];  val *= scale;  val += resid[m][n];
// Block tile 128x128x64, 8 waves (2x4), wave tile 64x32.
// Staging: GLOBAL_LOAD_ASYNC_TO_LDS_B128 (ASYNCcnt) into double-buffered LDS;
// 16 v_wmma_f32_16x16x32_bf16 per k-iteration, one barrier per iteration.
// ---------------------------------------------------------------------------
#define BM 128
#define BN 128
#define BK 64
#define LDP (BK + 8)   // 144B row stride: 16B-aligned, conflict-avoiding

__global__ __launch_bounds__(256) void k_gemm_bf16(
    const bf16_t* __restrict__ A, const bf16_t* __restrict__ BT,
    const float* __restrict__ bias, const float* __restrict__ resid,
    const float* __restrict__ scaleP, float* __restrict__ C,
    int M, int N, int K)
{
  __shared__ bf16_t As[2][BM][LDP];
  __shared__ bf16_t Bs[2][BN][LDP];

  const int tid  = threadIdx.x;
  const int lane = tid & 31;
  const int wave = tid >> 5;
  const int wm   = wave & 1;   // 2 wave-rows of 64
  const int wn   = wave >> 1;  // 4 wave-cols of 32
  const int m0   = blockIdx.y * BM;
  const int n0   = blockIdx.x * BN;

  // staging: 128 rows x 64 k x 2B = 16KB per matrix = 1024 16B-chunks;
  // 4 chunks per thread per matrix
  int rowL[4], koffL[4];
#pragma unroll
  for (int i = 0; i < 4; ++i) {
    const int c = tid + 256 * i;
    rowL[i]  = c >> 3;
    koffL[i] = (c & 7) * 8;   // element offset within the 64-wide k tile
  }

  auto stage = [&](int kt, int buf) {
    const int k0 = kt * BK;
#pragma unroll
    for (int i = 0; i < 4; ++i) {
      lds_copy16(A + (size_t)(m0 + rowL[i]) * K + k0 + koffL[i],
                 &As[buf][rowL[i]][koffL[i]]);
      lds_copy16(BT + (size_t)(n0 + rowL[i]) * K + k0 + koffL[i],
                 &Bs[buf][rowL[i]][koffL[i]]);
    }
  };

  v8f acc[4][2] = {};
  const int NKT = K / BK;

  stage(0, 0);
  wait_stage();
  __syncthreads();

  const int lr   = lane & 15;
  const int koff = (lane < 16) ? 0 : 8;   // K 0..7/16..23 vs 8..15/24..31

  for (int kt = 0; kt < NKT; ++kt) {
    const int buf = kt & 1;
    if (kt + 1 < NKT) stage(kt + 1, buf ^ 1);   // async: no VGPR residency

#pragma unroll
    for (int ks = 0; ks < 2; ++ks) {            // two 32-deep WMMA k-steps
      const int kb = ks * 32 + koff;
      v16bf afrag[4], bfrag[2];
#pragma unroll
      for (int mi = 0; mi < 4; ++mi) {
        const bf16_t* p = &As[buf][wm * 64 + mi * 16 + lr][kb];
        V16U u;
        u.h[0] = *(const v8bf*)p;
        u.h[1] = *(const v8bf*)(p + 16);
        afrag[mi] = u.v;
      }
#pragma unroll
      for (int ni = 0; ni < 2; ++ni) {
        const bf16_t* p = &Bs[buf][wn * 32 + ni * 16 + lr][kb];
        V16U u;
        u.h[0] = *(const v8bf*)p;
        u.h[1] = *(const v8bf*)(p + 16);
        bfrag[ni] = u.v;
      }
#pragma unroll
      for (int mi = 0; mi < 4; ++mi)
#pragma unroll
        for (int ni = 0; ni < 2; ++ni)
          acc[mi][ni] = __builtin_amdgcn_wmma_f32_16x16x32_bf16(
              false, afrag[mi], false, bfrag[ni], (short)0, acc[mi][ni],
              false, false);
    }

    if (kt + 1 < NKT) {
      wait_stage();        // next tile fully resident in LDS (this wave's part)
      __syncthreads();     // everyone's part resident; prior reads done
    }
  }

  const float scl = scaleP ? scaleP[0] : 1.0f;
#pragma unroll
  for (int mi = 0; mi < 4; ++mi) {
#pragma unroll
    for (int ni = 0; ni < 2; ++ni) {
      const int nn = n0 + wn * 32 + ni * 16 + lr;
      const int mb = m0 + wm * 64 + mi * 16 + ((lane < 16) ? 0 : 8);
      const float bvv = bias ? bias[nn] : 0.f;
#pragma unroll
      for (int r = 0; r < 8; ++r) {
        const int mm = mb + r;
        float val = (acc[mi][ni][r] + bvv) * scl;
        if (resid) val += resid[(size_t)mm * N + nn];
        C[(size_t)mm * N + nn] = val;
      }
    }
  }
}

// ---------------------------------------------------------------------------
// Scan: y_s[b,d] = sum_t m_t,s ; m_t,s = a_s^t m_t,s-1 + (1-a_s^t) v_t,s[b,d]
// Chunked 3-pass parallel prefix over S (32 chunks of 64).
// ---------------------------------------------------------------------------
__global__ void k_acum(const float* __restrict__ alpha, float* __restrict__ acum)
{
  const int gid = blockIdx.x * 256 + threadIdx.x;  // 2*32*5 = 320
  if (gid >= Bb * NCH * Tt) return;
  const int t = gid % Tt;
  const int c = (gid / Tt) % NCH;
  const int b = gid / (Tt * NCH);
  float ac = 1.f;
  for (int j = 0; j < CL; ++j) {
    const int row = b * Ss + c * CL + j;
    ac *= alpha[(size_t)row * Tt + t];
    acum[(size_t)row * Tt + t] = ac;
  }
}

__global__ __launch_bounds__(256) void k_scan1(
    const float* __restrict__ V, const float* __restrict__ alpha,
    float* __restrict__ ylocal, float* __restrict__ Epart)
{
  const int blk  = blockIdx.x;         // 128 blocks: (b, dblk, chunk)
  const int c    = blk & 31;
  const int dblk = (blk >> 5) & 1;
  const int b    = blk >> 6;
  const int d    = dblk * 256 + threadIdx.x;

  __shared__ float as[CL][Tt];
  for (int i = threadIdx.x; i < CL * Tt; i += 256) {
    const int s = i / Tt, t = i % Tt;
    as[s][t] = alpha[(size_t)(b * Ss + c * CL + s) * Tt + t];
  }
  __syncthreads();

  float g[Tt] = {0.f, 0.f, 0.f, 0.f, 0.f};
  for (int s = 0; s < CL; ++s) {
    const int row = b * Ss + c * CL + s;
    const float* vr = V + (size_t)row * NV + d;
    float y = 0.f;
#pragma unroll
    for (int t = 0; t < Tt; ++t) {
      const float a = as[s][t];
      g[t] = fmaf(a, g[t], (1.f - a) * vr[t * Dd]);
      y += g[t];
    }
    ylocal[(size_t)row * Dd + d] = y;
  }
#pragma unroll
  for (int t = 0; t < Tt; ++t)
    Epart[((size_t)(b * Tt + t) * NCH + c) * Dd + d] = g[t];
}

__global__ void k_scan2(
    const float* __restrict__ Epart, const float* __restrict__ acum,
    const float* __restrict__ mix, const float* __restrict__ h_prev,
    float* __restrict__ Gc)
{
  const int gid = blockIdx.x * 256 + threadIdx.x;  // 2*5*512 = 5120
  const int d = gid & 511;
  const int t = (gid >> 9) % Tt;
  const int b = gid / (Dd * Tt);
  float G = 0.f;
#pragma unroll
  for (int p = 0; p < Pp; ++p)
    G = fmaf(mix[p * Tt + t],
             h_prev[(((size_t)b * Pp + p) * Tt + t) * Dd + d], G);
  for (int c = 0; c < NCH; ++c) {
    const size_t idx = ((size_t)(b * Tt + t) * NCH + c) * Dd + d;
    Gc[idx] = G;
    const float P = acum[(size_t)(b * Ss + c * CL + CL - 1) * Tt + t];
    G = fmaf(P, G, Epart[idx]);
  }
}

__global__ __launch_bounds__(256) void k_scan3(
    const float* __restrict__ ylocal, const float* __restrict__ acum,
    const float* __restrict__ Gc, bf16_t* __restrict__ ybf)
{
  const int gid = blockIdx.x * 256 + threadIdx.x;  // 4096*512
  const int d = gid & 511;
  const int row = gid >> 9;
  const int b = row >> 11;
  const int s = row & 2047;
  const int c = s >> 6;
  float y = ylocal[gid];
#pragma unroll
  for (int t = 0; t < Tt; ++t)
    y = fmaf(acum[(size_t)row * Tt + t],
             Gc[((size_t)(b * Tt + t) * NCH + c) * Dd + d], y);
  ybf[gid] = (bf16_t)y;
}

// ---------------------------------------------------------------------------
extern "C" void kernel_launch(void* const* d_in, const int* in_sizes, int n_in,
                              void* d_out, int out_size, void* d_ws,
                              size_t ws_size, hipStream_t stream)
{
  (void)in_sizes; (void)n_in; (void)out_size; (void)ws_size;
  const float* hidden  = (const float*)d_in[0];
  const float* gamma   = (const float*)d_in[1];
  const float* beta    = (const float*)d_in[2];
  const float* W_in    = (const float*)d_in[3];
  const float* b_in    = (const float*)d_in[4];
  const float* W_map1  = (const float*)d_in[5];
  const float* b_map1  = (const float*)d_in[6];
  const float* W_map2  = (const float*)d_in[7];
  const float* b_map2  = (const float*)d_in[8];
  const float* log_tau = (const float*)d_in[9];
  const float* mix     = (const float*)d_in[10];
  const float* W_out   = (const float*)d_in[11];
  const float* b_out   = (const float*)d_in[12];
  const float* scale   = (const float*)d_in[13];
  const float* h_prev  = (const float*)d_in[14];

  char* w = (char*)d_ws;
  size_t o = 0;
  auto carve = [&](size_t bytes) {
    void* p = w + o;
    o = (o + bytes + 255) & ~(size_t)255;
    return p;
  };
  bf16_t* xn_bf  = (bf16_t*)carve((size_t)NTOK * Dd * 2);
  bf16_t* WvT    = (bf16_t*)carve((size_t)NV * Dd * 2);
  float*  bv     = (float*) carve((size_t)NV * 4);
  float*  alpha  = (float*) carve((size_t)NTOK * Tt * 4);
  float*  acum   = (float*) carve((size_t)NTOK * Tt * 4);
  float*  V      = (float*) carve((size_t)NTOK * NV * 4);
  float*  ylocal = (float*) carve((size_t)NTOK * Dd * 4);
  float*  Epart  = (float*) carve((size_t)Bb * Tt * NCH * Dd * 4);
  float*  Gc     = (float*) carve((size_t)Bb * Tt * NCH * Dd * 4);
  bf16_t* ybf    = (bf16_t*)carve((size_t)NTOK * Dd * 2);
  bf16_t* WoutT  = (bf16_t*)carve((size_t)Dd * Dd * 2);

  // 1. LayerNorm + gate -> xn(bf16), alpha
  k_ln_gate<<<NTOK / 8, 256, 0, stream>>>(hidden, gamma, beta, W_map1, b_map1,
                                          W_map2, b_map2, log_tau, xn_bf, alpha);
  // 2. Fold mix over p into weights
  k_build_wvt<<<(Tt * Dd * Dd) / 256, 256, 0, stream>>>(W_in, b_in, mix, WvT, bv);
  k_build_woutT<<<(Dd * Dd) / 256, 256, 0, stream>>>(W_out, WoutT);
  // 3. alpha cumulative products per chunk
  k_acum<<<2, 256, 0, stream>>>(alpha, acum);
  // 4. V = xn @ [Wv_0 | ... | Wv_4] + bv   (WMMA bf16)
  {
    dim3 grid(NV / BN, NTOK / BM);
    k_gemm_bf16<<<grid, 256, 0, stream>>>(xn_bf, WvT, bv, nullptr, nullptr, V,
                                          NTOK, NV, Dd);
  }
  // 5. Chunked linear scan
  k_scan1<<<Bb * 2 * NCH, 256, 0, stream>>>(V, alpha, ylocal, Epart);
  k_scan2<<<(Bb * Tt * Dd) / 256, 256, 0, stream>>>(Epart, acum, mix, h_prev, Gc);
  k_scan3<<<(NTOK * Dd) / 256, 256, 0, stream>>>(ylocal, acum, Gc, ybf);
  // 6. out = hidden + scale * (y @ W_out + b_out)   (WMMA bf16)
  {
    dim3 grid(Dd / BN, NTOK / BM);
    k_gemm_bf16<<<grid, 256, 0, stream>>>(ybf, WoutT, b_out, hidden, scale,
                                          (float*)d_out, NTOK, Dd, Dd);
  }
}
